// LookupTableMy_83726092468325
// MI455X (gfx1250) — matrix operand order repaired and checked by reading, not compiled, not run on hardware
//
#include <hip/hip_runtime.h>
#include <math.h>

typedef __attribute__((ext_vector_type(16))) __bf16 v16bf;
typedef __attribute__((ext_vector_type(8)))  __bf16 v8bf;
typedef __attribute__((ext_vector_type(4)))  __bf16 v4bf;
typedef __attribute__((ext_vector_type(8)))  float  v8f;
typedef __attribute__((ext_vector_type(4)))  float  f32x4;

#define BATCH 4096
#define DIN   128
#define HID   384
#define G4    1536   // 4*HID
#define L1    19
#define L2    20
#define NWIN  18

#define MT    32                 // batch rows per block (2 WMMA M-tiles)
#define XPAD  136                // 128 + 8 bf16 pad  (272B row, 68 dw == 4 mod 64 banks)
#define HPAD  392                // 384 + 8 bf16 pad  (784B row, 196 dw == 4 mod 64 banks)

// ---- async global->LDS copy (CDNA5 GLOBAL_LOAD_ASYNC_TO_LDS_B128), guarded ----
#if defined(__has_builtin)
#if __has_builtin(__builtin_amdgcn_global_load_async_to_lds_b128) && \
    __has_builtin(__builtin_amdgcn_s_wait_asynccnt)
#define USE_ASYNC_LDS 1
#endif
#endif
#ifndef USE_ASYNC_LDS
#define USE_ASYNC_LDS 0
#endif

// GCC-style int4 vector: matches the builtin's parameter type
// 'int __attribute__((vector_size(16))) __device__ *'.
typedef int v4i_g __attribute__((vector_size(16)));
typedef __attribute__((address_space(1))) v4i_g* as1_v4i;
typedef __attribute__((address_space(3))) v4i_g* as3_v4i;

__device__ __forceinline__ void copy16_g2l(const void* g, void* l) {
#if USE_ASYNC_LDS
  __builtin_amdgcn_global_load_async_to_lds_b128((as1_v4i)g, (as3_v4i)l, 0, 0);
#else
  *(uint4*)l = *(const uint4*)g;
#endif
}
__device__ __forceinline__ void async_fence() {
#if USE_ASYNC_LDS
  __builtin_amdgcn_s_wait_asynccnt(0);
#endif
}

// ---- fast gate math on hardware TRANS ops (v_exp_f32 / v_rcp_f32, ~1 ulp) ----
#define LOG2E_F  1.4426950408889634f
__device__ __forceinline__ float fast_sigmoid(float x) {
  // 1/(1+e^-x); x->+inf: exp2->0 -> 1; x->-inf: exp2->inf -> rcp->0
  float e = __builtin_amdgcn_exp2f(-LOG2E_F * x);
  return __builtin_amdgcn_rcpf(1.0f + e);
}
__device__ __forceinline__ float fast_tanh(float x) {
  // 1 - 2/(e^{2x}+1); saturates correctly to +/-1 via inf/rcp semantics
  float e = __builtin_amdgcn_exp2f(2.0f * LOG2E_F * x);
  return 1.0f - 2.0f * __builtin_amdgcn_rcpf(e + 1.0f);
}

// ---------------- prep: bf16 weights, fused bias, zero c ----------------
__global__ void lstm_prep(const float* __restrict__ W_ih, const float* __restrict__ W_hh,
                          const float* __restrict__ b_ih, const float* __restrict__ b_hh,
                          __bf16* __restrict__ Wih_bf, __bf16* __restrict__ Whh_bf,
                          float* __restrict__ bias, float* __restrict__ c) {
  int i = blockIdx.x * blockDim.x + threadIdx.x;
  if (i < G4 * DIN)   Wih_bf[i] = (__bf16)W_ih[i];
  if (i < G4 * HID)   Whh_bf[i] = (__bf16)W_hh[i];
  if (i < G4)         bias[i]   = b_ih[i] + b_hh[i];
  if (i < BATCH*HID)  c[i]      = 0.0f;
}

// ---------------- gather embeddings for all 19 steps into bf16 X ----------------
__global__ void gather_x(const int* __restrict__ input1, const float* __restrict__ emb,
                         __bf16* __restrict__ X) {
  int e = blockIdx.x * blockDim.x + threadIdx.x;      // over L1*BATCH*32 quads
  if (e >= L1 * BATCH * 32) return;
  int kq = e & 31;
  int m  = (e >> 5) & (BATCH - 1);
  int t  = e >> 17;                                   // BATCH*32 == 2^17
  int idx = input1[m * L1 + t];
  f32x4 v = *(const f32x4*)(emb + (size_t)idx * DIN + kq * 4);
  v4bf o;
  o[0] = (__bf16)v[0]; o[1] = (__bf16)v[1]; o[2] = (__bf16)v[2]; o[3] = (__bf16)v[3];
  *(v4bf*)(X + ((size_t)t * BATCH + m) * DIN + kq * 4) = o;
}

// ---------------- fragment helpers ----------------
__device__ __forceinline__ v16bf combine(v8bf lo, v8bf hi) {
  v16bf a;
  #pragma unroll
  for (int j = 0; j < 8; ++j) { a[j] = lo[j]; a[j + 8] = hi[j]; }
  return a;
}
// A fragment (16x32 bf16) from an LDS tile: lane holds row M = lane&15,
// element j -> K = (lane>=16 ? 8:0) + (j<8 ? j : j+8)
__device__ __forceinline__ v16bf frag_a_lds(const __bf16* tile, int stride, int k0, int lane) {
  const __bf16* p = tile + (size_t)(lane & 15) * stride + k0 + ((lane & 16) ? 8 : 0);
  return combine(*(const v8bf*)p, *(const v8bf*)(p + 16));
}
// B fragment (32x16 bf16), B[k][n] = W[n][k]: lane gives N, element j -> K = (lane>=16?8:0)+(j<8?j:j+8)
__device__ __forceinline__ v16bf frag_b(const __bf16* __restrict__ W, int ld,
                                        int n, int k0, int lane) {
  const __bf16* p = W + (size_t)n * ld + k0 + ((lane & 16) ? 8 : 0);
  return combine(*(const v8bf*)p, *(const v8bf*)(p + 16));
}

#define WMMA_BF16(A, B, C) \
  __builtin_amdgcn_wmma_f32_16x16x32_bf16(false, (A), false, (B), (short)0, (C), false, false)

// ---------------- one LSTM step ----------------
// grid = (BATCH/MT, 3): x = M-block (32 rows), y = N-group (one 16-col tile per gate
// per wave). block = 256 threads (8 waves). Ping-pong h buffers kill all h hazards.
__global__ void __launch_bounds__(256)
lstm_step(const __bf16* __restrict__ Xt,     // (BATCH, DIN) bf16, this step's inputs
          const __bf16* __restrict__ hprev,  // (BATCH, HID) bf16
          __bf16*       __restrict__ hnext,  // (BATCH, HID) bf16
          float*        __restrict__ c,      // (BATCH, HID) f32
          float*        __restrict__ h32,    // (BATCH, HID) f32 (written on last step)
          const __bf16* __restrict__ Wih, const __bf16* __restrict__ Whh,
          const float*  __restrict__ bias, int skip_h, int store_h32) {
  __shared__ __attribute__((aligned(32))) __bf16 lds_x[MT * XPAD];
  __shared__ __attribute__((aligned(32))) __bf16 lds_h[MT * HPAD];

  const int lane = threadIdx.x & 31;
  const int wv   = threadIdx.x >> 5;       // 0..7
  const int ngrp = blockIdx.y;             // 0..2
  const int row0 = blockIdx.x * MT;

  // ---- cooperative stage of A tiles into LDS (async DMA engine) ----
  {
    // x tile: 32 rows x 128 bf16 = 512 16B-chunks
    #pragma unroll
    for (int i = 0; i < 2; ++i) {
      int ch = threadIdx.x + i * 256;            // 0..511
      int r = ch >> 4, cc = ch & 15;
      copy16_g2l(Xt + (size_t)(row0 + r) * DIN + cc * 8, lds_x + (size_t)r * XPAD + cc * 8);
    }
    if (!skip_h) {
      // h tile: 32 rows x 384 bf16 = 1536 16B-chunks
      #pragma unroll
      for (int i = 0; i < 6; ++i) {
        int ch = threadIdx.x + i * 256;          // 0..1535
        int r = ch / 48, cc = ch % 48;
        copy16_g2l(hprev + (size_t)(row0 + r) * HID + cc * 8, lds_h + (size_t)r * HPAD + cc * 8);
      }
    }
    async_fence();
    __syncthreads();
  }

  v8f acc[4][2];
  #pragma unroll
  for (int g = 0; g < 4; ++g) { acc[g][0] = (v8f)0.0f; acc[g][1] = (v8f)0.0f; }

  const int ncol = (ngrp * 8 + wv) * 16 + (lane & 15);   // hidden unit 0..383

  // ---- input projection: K = 128 (4 k-tiles) ----
  #pragma unroll
  for (int kt = 0; kt < 4; ++kt) {
    v16bf a0 = frag_a_lds(lds_x,             XPAD, kt * 32, lane);
    v16bf a1 = frag_a_lds(lds_x + 16 * XPAD, XPAD, kt * 32, lane);
    #pragma unroll
    for (int g = 0; g < 4; ++g) {
      v16bf b = frag_b(Wih, DIN, g * HID + ncol, kt * 32, lane);
      acc[g][0] = WMMA_BF16(a0, b, acc[g][0]);
      acc[g][1] = WMMA_BF16(a1, b, acc[g][1]);
    }
  }

  // ---- recurrent projection: K = 384 (12 k-tiles) ----
  if (!skip_h) {
    #pragma unroll 4
    for (int kt = 0; kt < 12; ++kt) {
      v16bf a0 = frag_a_lds(lds_h,             HPAD, kt * 32, lane);
      v16bf a1 = frag_a_lds(lds_h + 16 * HPAD, HPAD, kt * 32, lane);
      #pragma unroll
      for (int g = 0; g < 4; ++g) {
        v16bf b = frag_b(Whh, HID, g * HID + ncol, kt * 32, lane);
        acc[g][0] = WMMA_BF16(a0, b, acc[g][0]);
        acc[g][1] = WMMA_BF16(a1, b, acc[g][1]);
      }
    }
  }

  // ---- gates + state update (i/f/g/o fragments all live in this wave) ----
  const float bi = bias[ncol],           bf2 = bias[HID + ncol];
  const float bg = bias[2*HID + ncol],   bo  = bias[3*HID + ncol];
  #pragma unroll
  for (int mt = 0; mt < 2; ++mt) {
    #pragma unroll
    for (int r = 0; r < 8; ++r) {
      const int row = row0 + mt * 16 + r + ((lane & 16) ? 8 : 0);
      const size_t o = (size_t)row * HID + ncol;
      float ig = fast_sigmoid(acc[0][mt][r] + bi);
      float fg = fast_sigmoid(acc[1][mt][r] + bf2);
      float gg = fast_tanh   (acc[2][mt][r] + bg);
      float og = fast_sigmoid(acc[3][mt][r] + bo);
      float cn = fg * c[o] + ig * gg;
      float hn = og * fast_tanh(cn);
      c[o] = cn;
      hnext[o] = (__bf16)hn;
      if (store_h32) h32[o] = hn;
    }
  }
}

// ---------------- windows + max + log_softmax, one wave per batch row ----------------
__global__ void __launch_bounds__(32)
finalize(const int* __restrict__ input2, const float* __restrict__ emb,
         const float* __restrict__ h, const float* __restrict__ lin_w,
         const float* __restrict__ lin_b, float* __restrict__ out) {
  const int m = blockIdx.x;
  const int lane = threadIdx.x;
  const float* hrow = h + (size_t)m * HID;

  float ms = -INFINITY;
  for (int n = 0; n < NWIN; ++n) {
    const int i0 = input2[m * L2 + n + 0];
    const int i1 = input2[m * L2 + n + 1];
    const int i2 = input2[m * L2 + n + 2];
    float p = 0.0f;
    for (int k = lane; k < HID; k += 32) {
      const int w = k >> 7;
      const int tok = (w == 0) ? i0 : (w == 1) ? i1 : i2;
      p += hrow[k] * emb[(size_t)tok * DIN + (k & 127)];
    }
    #pragma unroll
    for (int s = 16; s > 0; s >>= 1) p += __shfl_xor(p, s, 32);
    ms = fmaxf(ms, p);
  }
  if (lane == 0) {
    float l0 = ms * lin_w[0] + lin_b[0];
    float l1 = ms * lin_w[1] + lin_b[1];
    float mx = fmaxf(l0, l1);
    float lse = mx + logf(expf(l0 - mx) + expf(l1 - mx));
    out[m * 2 + 0] = l0 - lse;
    out[m * 2 + 1] = l1 - lse;
  }
}

// ---------------- launcher ----------------
extern "C" void kernel_launch(void* const* d_in, const int* in_sizes, int n_in,
                              void* d_out, int out_size, void* d_ws, size_t ws_size,
                              hipStream_t stream) {
  const int*   input1 = (const int*)  d_in[0];
  const int*   input2 = (const int*)  d_in[1];
  const float* emb    = (const float*)d_in[2];
  const float* W_ih   = (const float*)d_in[3];
  const float* W_hh   = (const float*)d_in[4];
  const float* b_ih   = (const float*)d_in[5];
  const float* b_hh   = (const float*)d_in[6];
  const float* lin_w  = (const float*)d_in[7];
  const float* lin_b  = (const float*)d_in[8];
  float* out = (float*)d_out;

  char* w = (char*)d_ws;
  float*  cst    = (float*)w;   w += (size_t)BATCH * HID * 4;     // 6.29 MB
  float*  h32    = (float*)w;   w += (size_t)BATCH * HID * 4;     // 6.29 MB
  __bf16* hbfA   = (__bf16*)w;  w += (size_t)BATCH * HID * 2;     // 3.15 MB
  __bf16* hbfB   = (__bf16*)w;  w += (size_t)BATCH * HID * 2;     // 3.15 MB
  __bf16* Wih_bf = (__bf16*)w;  w += (size_t)G4 * DIN * 2;        // 393 KB
  __bf16* Whh_bf = (__bf16*)w;  w += (size_t)G4 * HID * 2;        // 1.18 MB
  float*  bias   = (float*)w;   w += (size_t)G4 * 4;
  __bf16* X      = (__bf16*)w;  w += (size_t)L1 * BATCH * DIN * 2;// 19.9 MB

  lstm_prep<<<(BATCH * HID + 255) / 256, 256, 0, stream>>>(
      W_ih, W_hh, b_ih, b_hh, Wih_bf, Whh_bf, bias, cst);

  gather_x<<<(L1 * BATCH * 32 + 255) / 256, 256, 0, stream>>>(input1, emb, X);

  for (int t = 0; t < L1; ++t) {
    const __bf16* hp = (t & 1) ? hbfB : hbfA;   // read  (unused at t==0)
    __bf16*       hn = (t & 1) ? hbfA : hbfB;   // write
    lstm_step<<<dim3(BATCH / MT, 3), 256, 0, stream>>>(
        X + (size_t)t * BATCH * DIN, hp, hn, cst, h32,
        Wih_bf, Whh_bf, bias, t == 0 ? 1 : 0, t == L1 - 1 ? 1 : 0);
  }

  finalize<<<BATCH, 32, 0, stream>>>(input2, emb, h32, lin_w, lin_b, out);
}
